// MyModel_86981677679366
// MI455X (gfx1250) — matrix operand order repaired and checked by reading, not compiled, not run on hardware
//
#include <hip/hip_runtime.h>
#include <cmath>

// ---------------- problem constants ----------------
constexpr int kN   = 60000;   // nodes per ntype
constexpr int kE   = 600000;  // edges per relation
constexpr int kD   = 128;     // hidden dim
constexpr int kDin = 256;     // input dim
constexpr int kDff = 2048;    // FFN dim
constexpr int kOut = 16;      // output dim

typedef __attribute__((ext_vector_type(8)))  _Float16 v8h;
typedef __attribute__((ext_vector_type(16))) _Float16 v16h;
typedef __attribute__((ext_vector_type(8)))  float    v8f;

#define DEVI static __device__ __forceinline__

// ---------------- WMMA fragment helpers ----------------
// A frag (16x32 f16): lane L holds row M=L&15; kg=L>>4; halves e0..7 -> K=kg*8+e,
// e8..15 -> K=kg*8+16+(e-8).  B frag (32x16): lane L holds col N=L&15; same
// K order as A.  C/D (v8f): lane -> col N=L&15, vgpr r -> row M=kg*8+r.
DEVI v16h frag_cat(v8h lo, v8h hi) {
  v16h r;
#pragma unroll
  for (int i = 0; i < 8; i++) { r[i] = lo[i]; r[i + 8] = hi[i]; }
  return r;
}
DEVI v16h load_afrag_f16(const _Float16* p) {
  v8h lo = *(const v8h*)p;
  v8h hi = *(const v8h*)(p + 16);
  return frag_cat(lo, hi);
}
DEVI v16h load_bfrag(const _Float16* p) {
  v8h lo = *(const v8h*)p;
  v8h hi = *(const v8h*)(p + 16);
  return frag_cat(lo, hi);
}
// f32 A-side load with optional fused prologue: a = (A + R) * S[row], cvt to f16
DEVI v16h load_afrag_f32(const float* A, const float* R, const float* S,
                         int row, int lda, int kk) {
  const float4* p4 = (const float4*)(A + (size_t)row * lda + kk);
  float4 x0 = p4[0], x1 = p4[1], x2 = p4[4], x3 = p4[5];
  float f[16] = {x0.x, x0.y, x0.z, x0.w, x1.x, x1.y, x1.z, x1.w,
                 x2.x, x2.y, x2.z, x2.w, x3.x, x3.y, x3.z, x3.w};
  if (R) {
    const float4* q4 = (const float4*)(R + (size_t)row * lda + kk);
    float4 y0 = q4[0], y1 = q4[1], y2 = q4[4], y3 = q4[5];
    float g[16] = {y0.x, y0.y, y0.z, y0.w, y1.x, y1.y, y1.z, y1.w,
                   y2.x, y2.y, y2.z, y2.w, y3.x, y3.y, y3.z, y3.w};
#pragma unroll
    for (int i = 0; i < 16; i++) f[i] += g[i];
  }
  if (S) {
    float s = S[row];
#pragma unroll
    for (int i = 0; i < 16; i++) f[i] *= s;
  }
  v16h r;
#pragma unroll
  for (int i = 0; i < 16; i++) r[i] = (_Float16)f[i];
  return r;
}
DEVI v8f wmma_f16(v16h a, v16h b, v8f c) {
  return __builtin_amdgcn_wmma_f32_16x16x32_f16(false, a, false, b, (short)0, c,
                                                false, false);
}

// ---------------- generic NT GEMM:  C = act(A @ W^T + bias) ----------------
// A: (M,K) f16 or f32 (with optional (A+R)*S prologue), W: (Nc,K) f16 row-major.
// Wave computes a 32x32 tile; block = 8 waves = 64 rows x 128 cols.
// act: 0=none 1=PReLU(alphaPtr[0]) 2=ReLU 3=tanh
template <bool AF16>
__global__ void __launch_bounds__(256)
gemm_nt(const void* __restrict__ Ap, int lda,
        const _Float16* __restrict__ W, int ldw,
        const float* __restrict__ bias,
        const float* __restrict__ R, const float* __restrict__ S,
        const float* __restrict__ alphaPtr, int act,
        float* __restrict__ C, int ldc,
        _Float16* __restrict__ C16, int ldc16,
        int M, int Nc, int K) {
  const int lane = threadIdx.x;
  const int wv = threadIdx.y;
  const int wm = wv >> 2, wn = wv & 3;
  const int row0 = blockIdx.x * 64 + wm * 32;
  const int col0 = blockIdx.y * 128 + wn * 32;
  if (row0 >= M || col0 >= Nc) return;   // wave-uniform exit (EXEC stays all-1)
  const int l15 = lane & 15, kg = lane >> 4;
  const int mA0 = (row0 + l15 < M) ? row0 + l15 : M - 1;
  const int mA1 = (row0 + 16 + l15 < M) ? row0 + 16 + l15 : M - 1;
  const int nB0 = (col0 + l15 < Nc) ? col0 + l15 : Nc - 1;
  const int nB1 = (col0 + 16 + l15 < Nc) ? col0 + 16 + l15 : Nc - 1;
  v8f c00 = {}, c01 = {}, c10 = {}, c11 = {};
  for (int k0 = 0; k0 < K; k0 += 32) {
    v16h a0, a1;
    if (AF16) {
      const _Float16* A = (const _Float16*)Ap;
      a0 = load_afrag_f16(A + (size_t)mA0 * lda + k0 + kg * 8);
      a1 = load_afrag_f16(A + (size_t)mA1 * lda + k0 + kg * 8);
    } else {
      const float* A = (const float*)Ap;
      a0 = load_afrag_f32(A, R, S, mA0, lda, k0 + kg * 8);
      a1 = load_afrag_f32(A, R, S, mA1, lda, k0 + kg * 8);
    }
    v16h b0 = load_bfrag(W + (size_t)nB0 * ldw + k0 + kg * 8);
    v16h b1 = load_bfrag(W + (size_t)nB1 * ldw + k0 + kg * 8);
    c00 = wmma_f16(a0, b0, c00);
    c01 = wmma_f16(a0, b1, c01);
    c10 = wmma_f16(a1, b0, c10);
    c11 = wmma_f16(a1, b1, c11);
  }
  const float alpha = (act == 1) ? alphaPtr[0] : 0.f;
  auto ep = [&](v8f cf, int rb, int cb) {
    int n = cb + l15;
    bool nok = n < Nc;
    float bv = (bias && nok) ? bias[n] : 0.f;
#pragma unroll
    for (int r = 0; r < 8; r++) {
      int m = rb + kg * 8 + r;
      float v = cf[r] + bv;
      if (act == 1) v = (v >= 0.f) ? v : alpha * v;
      else if (act == 2) v = (v > 0.f) ? v : 0.f;
      else if (act == 3) v = tanhf(v);
      if (m < M && nok) {
        if (C)   C[(size_t)m * ldc + n] = v;
        if (C16) C16[(size_t)m * ldc16 + n] = (_Float16)v;
      }
    }
  };
  ep(c00, row0, col0);
  ep(c01, row0, col0 + 16);
  ep(c10, row0 + 16, col0);
  ep(c11, row0 + 16, col0 + 16);
}

// ---------------- fused FFN: Out = relu(X @ W1^T + b1) @ W2^T + b2 ----------
// X:(M,128) f16, W1:(2048,128) f16, W2:(128,2048) f16, Out:(M,128) f32.
// Wave owns 16 rows; hidden processed in chunks of 32; intermediate relu tile
// goes through per-wave LDS to convert C-layout (f32) -> A-fragment (f16).
__global__ void __launch_bounds__(256)
k_ffn(const _Float16* __restrict__ X, const _Float16* __restrict__ W1,
      const float* __restrict__ b1, const _Float16* __restrict__ W2,
      const float* __restrict__ b2, float* __restrict__ Out, int M) {
  __shared__ __align__(16) _Float16 lds[8][16][32];
  const int lane = threadIdx.x, wv = threadIdx.y;
  const int r0 = blockIdx.x * 128 + wv * 16;
  if (r0 >= M) return;
  const int l15 = lane & 15, kg = lane >> 4;
  const int mA = (r0 + l15 < M) ? r0 + l15 : M - 1;
  v16h ax[4];
#pragma unroll
  for (int kc = 0; kc < 4; kc++)
    ax[kc] = load_afrag_f16(X + (size_t)mA * kD + kc * 32 + kg * 8);
  v8f zero = {};
  v8f acc[8];
#pragma unroll
  for (int t = 0; t < 8; t++) acc[t] = zero;
  const int jl0 = l15, jl1 = 16 + l15;
  for (int j0 = 0; j0 < kDff; j0 += 32) {
    v8f h0 = {}, h1 = {};
#pragma unroll
    for (int kc = 0; kc < 4; kc++) {
      v16h wb0 = load_bfrag(W1 + (size_t)(j0 + jl0) * kD + kc * 32 + kg * 8);
      v16h wb1 = load_bfrag(W1 + (size_t)(j0 + jl1) * kD + kc * 32 + kg * 8);
      h0 = wmma_f16(ax[kc], wb0, h0);
      h1 = wmma_f16(ax[kc], wb1, h1);
    }
    float bb0 = b1[j0 + jl0], bb1 = b1[j0 + jl1];
    int mr = kg * 8;
#pragma unroll
    for (int rr = 0; rr < 8; rr++) {
      float u0 = h0[rr] + bb0; u0 = (u0 > 0.f) ? u0 : 0.f;
      float u1 = h1[rr] + bb1; u1 = (u1 > 0.f) ? u1 : 0.f;
      lds[wv][mr + rr][jl0] = (_Float16)u0;
      lds[wv][mr + rr][jl1] = (_Float16)u1;
    }
    // wave-private LDS exchange: drain DS stores before cross-lane reads
    asm volatile("s_wait_dscnt 0x0" ::: "memory");
    v16h ah = load_afrag_f16(&lds[wv][l15][kg * 8]);
#pragma unroll
    for (int t = 0; t < 8; t++) {
      v16h wb = load_bfrag(W2 + (size_t)(t * 16 + l15) * kDff + j0 + kg * 8);
      acc[t] = wmma_f16(ah, wb, acc[t]);
    }
    // drain DS loads before next iteration overwrites the tile
    asm volatile("s_wait_dscnt 0x0" ::: "memory");
  }
#pragma unroll
  for (int t = 0; t < 8; t++) {
    int n = t * 16 + l15;
    float bv = b2[n];
#pragma unroll
    for (int rr = 0; rr < 8; rr++) {
      int m = r0 + kg * 8 + rr;
      if (m < M) Out[(size_t)m * kD + n] = acc[t][rr] + bv;
    }
  }
}

// ---------------- small helper kernels ----------------
__global__ void k_cvt16(const float* __restrict__ in, _Float16* __restrict__ out,
                        long long n) {
  long long i = (long long)blockIdx.x * blockDim.x + threadIdx.x;
  if (i < n) out[i] = (_Float16)in[i];
}
// out[c][r] = in[r][c]  (store W^T so x@W becomes the NT GEMM form)
__global__ void k_cvt16_t(const float* __restrict__ in, _Float16* __restrict__ out,
                          int rows, int cols) {
  int i = blockIdx.x * blockDim.x + threadIdx.x;
  if (i < rows * cols) {
    int r = i / cols, c = i % cols;
    out[(size_t)c * rows + r] = (_Float16)in[i];
  }
}
__global__ void k_count(const int* __restrict__ dst, float* __restrict__ deg, int E) {
  int i = blockIdx.x * blockDim.x + threadIdx.x;
  if (i < E) atomicAdd(&deg[dst[i]], 1.f);
}
__global__ void k_invdeg(float* deg, int n) {
  int i = blockIdx.x * blockDim.x + threadIdx.x;
  if (i < n) deg[i] = 1.f / (deg[i] + 1.f);
}
// msg[dst] += h[src]; 32 threads per edge, 4 floats each (coalesced 512B rows)
__global__ void k_scatter(const float* __restrict__ h, const int* __restrict__ src,
                          const int* __restrict__ dst, float* __restrict__ outm,
                          int E) {
  long long t = (long long)blockIdx.x * blockDim.x + threadIdx.x;
  int e = (int)(t >> 5);
  if (e >= E) return;
  int c = ((int)t & 31) * 4;
  const float4 v = *(const float4*)(h + (size_t)src[e] * kD + c);
  float* o = outm + (size_t)dst[e] * kD + c;
  atomicAdd(o + 0, v.x); atomicAdd(o + 1, v.y);
  atomicAdd(o + 2, v.z); atomicAdd(o + 3, v.w);
}
__global__ void k_colsum(const float* __restrict__ t, float* __restrict__ outc,
                         int rows) {
  int d = threadIdx.x;                 // 128 threads = 128 columns
  long long r0 = (long long)blockIdx.x * 256;
  float s = 0.f;
  for (int i = 0; i < 256; i++) {
    long long r = r0 + i;
    if (r < rows) s += t[r * kD + d];
  }
  atomicAdd(&outc[d], s);
}
__global__ void k_beta(const float* __restrict__ colsum, const float* __restrict__ q,
                       float* __restrict__ beta) {
  for (int tt = 0; tt < 2; tt++) {
    float w0 = 0.f, w1 = 0.f;
    for (int d = 0; d < kD; d++) {
      w0 += colsum[(tt * 2 + 0) * kD + d] * q[tt * kD + d];
      w1 += colsum[(tt * 2 + 1) * kD + d] * q[tt * kD + d];
    }
    w0 /= (float)kN; w1 /= (float)kN;
    float mx = (w0 > w1) ? w0 : w1;
    float e0 = __expf(w0 - mx), e1 = __expf(w1 - mx);
    float inv = 1.f / (e0 + e1);
    beta[tt * 2 + 0] = e0 * inv;
    beta[tt * 2 + 1] = e1 * inv;
  }
}
__global__ void k_combine(const float* __restrict__ h0, const float* __restrict__ h1,
                          const float* __restrict__ beta, float* __restrict__ tgt,
                          _Float16* __restrict__ tgt16, int slot) {
  long long i = (long long)blockIdx.x * blockDim.x + threadIdx.x;
  if (i >= (long long)kN * kD) return;
  float v = beta[0] * h0[i] + beta[1] * h1[i];
  long long n = i >> 7;
  int d = (int)(i & 127);
  size_t o = ((size_t)n * 2 + slot) * kD + d;
  tgt[o] = v;
  tgt16[o] = (_Float16)v;
}
// 8-head MHA over seq len 2 (dh=16). one thread per (node, head).
__global__ void k_mha(const float* __restrict__ qb, const float* __restrict__ kb,
                      const float* __restrict__ vb, int ld,
                      _Float16* __restrict__ o, int nN) {
  int t = blockIdx.x * blockDim.x + threadIdx.x;
  if (t >= nN * 8) return;
  int n = t >> 3, h = t & 7;
  size_t r0 = (size_t)(n * 2) * ld, r1 = (size_t)(n * 2 + 1) * ld;
  int off = h * 16;
  float q0[16], q1[16], k0[16], k1[16], v0[16], v1[16];
#pragma unroll
  for (int e = 0; e < 16; e++) {
    q0[e] = qb[r0 + off + e]; q1[e] = qb[r1 + off + e];
    k0[e] = kb[r0 + off + e]; k1[e] = kb[r1 + off + e];
    v0[e] = vb[r0 + off + e]; v1[e] = vb[r1 + off + e];
  }
  float s00 = 0.f, s01 = 0.f, s10 = 0.f, s11 = 0.f;
#pragma unroll
  for (int e = 0; e < 16; e++) {
    s00 += q0[e] * k0[e]; s01 += q0[e] * k1[e];
    s10 += q1[e] * k0[e]; s11 += q1[e] * k1[e];
  }
  const float sc = 0.25f;  // 1/sqrt(16)
  s00 *= sc; s01 *= sc; s10 *= sc; s11 *= sc;
  float m0 = (s00 > s01) ? s00 : s01;
  float e00 = __expf(s00 - m0), e01 = __expf(s01 - m0);
  float i0 = 1.f / (e00 + e01);
  float p00 = e00 * i0, p01 = e01 * i0;
  float m1 = (s10 > s11) ? s10 : s11;
  float e10 = __expf(s10 - m1), e11 = __expf(s11 - m1);
  float i1 = 1.f / (e10 + e11);
  float p10 = e10 * i1, p11 = e11 * i1;
  size_t o0 = (size_t)(n * 2) * kD + off, o1 = (size_t)(n * 2 + 1) * kD + off;
#pragma unroll
  for (int e = 0; e < 16; e++) {
    o[o0 + e] = (_Float16)(p00 * v0[e] + p01 * v1[e]);
    o[o1 + e] = (_Float16)(p10 * v0[e] + p11 * v1[e]);
  }
}
// LayerNorm over rows of 128: out = LN(a+r)*g+b ; one wave per row
__global__ void k_ln(const float* __restrict__ a, const float* __restrict__ r,
                     const float* __restrict__ g, const float* __restrict__ bb,
                     float* __restrict__ o32, _Float16* __restrict__ o16, int rows) {
  int row = blockIdx.x * 8 + threadIdx.y;
  if (row >= rows) return;
  int lane = threadIdx.x;
  const float* pa = a + (size_t)row * kD;
  const float* pr = r + (size_t)row * kD;
  float v[4];
  float s = 0.f;
#pragma unroll
  for (int i = 0; i < 4; i++) {
    int d = lane + 32 * i;
    v[i] = pa[d] + pr[d];
    s += v[i];
  }
  for (int m = 16; m >= 1; m >>= 1) s += __shfl_xor(s, m, 32);
  float mean = s * (1.f / kD);
  float var = 0.f;
#pragma unroll
  for (int i = 0; i < 4; i++) { float dd = v[i] - mean; var += dd * dd; }
  for (int m = 16; m >= 1; m >>= 1) var += __shfl_xor(var, m, 32);
  var *= (1.f / kD);
  float rstd = rsqrtf(var + 1e-5f);
#pragma unroll
  for (int i = 0; i < 4; i++) {
    int d = lane + 32 * i;
    float ov = (v[i] - mean) * rstd * g[d] + bb[d];
    if (o32) o32[(size_t)row * kD + d] = ov;
    if (o16) o16[(size_t)row * kD + d] = (_Float16)ov;
  }
}

// ---------------- host-side launch helpers ----------------
static inline void launch_gemm(hipStream_t s, bool af16, const void* A, int lda,
                               const _Float16* W, int ldw, const float* bias,
                               const float* R, const float* S, const float* alpha,
                               int act, float* C, int ldc, _Float16* C16, int ldc16,
                               int M, int Nc, int K) {
  dim3 blk(32, 8), grd((M + 63) / 64, (Nc + 127) / 128);
  if (af16)
    gemm_nt<true><<<grd, blk, 0, s>>>(A, lda, W, ldw, bias, R, S, alpha, act, C,
                                      ldc, C16, ldc16, M, Nc, K);
  else
    gemm_nt<false><<<grd, blk, 0, s>>>(A, lda, W, ldw, bias, R, S, alpha, act, C,
                                       ldc, C16, ldc16, M, Nc, K);
}
static inline void cvt(hipStream_t s, const float* in, _Float16* out, long long n) {
  k_cvt16<<<(unsigned)((n + 255) / 256), 256, 0, s>>>(in, out, n);
}
static inline void cvtT(hipStream_t s, const float* in, _Float16* out, int rows,
                        int cols) {
  k_cvt16_t<<<(rows * cols + 255) / 256, 256, 0, s>>>(in, out, rows, cols);
}

extern "C" void kernel_launch(void* const* d_in, const int* in_sizes, int n_in,
                              void* d_out, int out_size, void* d_ws, size_t ws_size,
                              hipStream_t stream) {
  (void)in_sizes; (void)n_in; (void)out_size; (void)ws_size;
  const float* feat_A    = (const float*)d_in[0];
  const float* feat_P    = (const float*)d_in[1];
  const float* rand_f    = (const float*)d_in[2];
  const float* fcA_w     = (const float*)d_in[3];
  const float* fcA_b     = (const float*)d_in[4];
  const float* fcP_w     = (const float*)d_in[5];
  const float* fcP_b     = (const float*)d_in[6];
  const float* in_w_pa   = (const float*)d_in[7];
  const float* in_w_ap   = (const float*)d_in[8];
  const float* gc_w      = (const float*)d_in[9];
  const float* gc_b      = (const float*)d_in[10];
  const float* prelu_a   = (const float*)d_in[11];
  const float* attn_fc_w = (const float*)d_in[12];
  const float* attn_fc_b = (const float*)d_in[13];
  const float* attn_vec  = (const float*)d_in[14];
  const float* sa_in_w   = (const float*)d_in[15];
  const float* sa_in_b   = (const float*)d_in[16];
  const float* sa_out_w  = (const float*)d_in[17];
  const float* sa_out_b  = (const float*)d_in[18];
  const float* ca_in_w   = (const float*)d_in[19];
  const float* ca_in_b   = (const float*)d_in[20];
  const float* ca_out_w  = (const float*)d_in[21];
  const float* ca_out_b  = (const float*)d_in[22];
  const float* ln_g      = (const float*)d_in[23];
  const float* ln_b      = (const float*)d_in[24];
  const float* ff_w1     = (const float*)d_in[25];
  const float* ff_b1     = (const float*)d_in[26];
  const float* ff_w2     = (const float*)d_in[27];
  const float* ff_b2     = (const float*)d_in[28];
  const float* pred_w    = (const float*)d_in[29];
  const float* pred_b    = (const float*)d_in[30];
  const int*   src_pa    = (const int*)d_in[31];
  const int*   dst_pa    = (const int*)d_in[32];
  const int*   src_ap    = (const int*)d_in[33];
  const int*   dst_ap    = (const int*)d_in[34];
  float* out = (float*)d_out;

  // -------- workspace bump allocator --------
  char* base = (char*)d_ws;
  size_t off = 0;
  auto alloc = [&](size_t bytes) {
    off = (off + 255) & ~(size_t)255;
    void* p = base + off;
    off += bytes;
    return p;
  };
  const size_t Nf = (size_t)kN * kD;       // elems per (N,128)
  const size_t Nf2 = (size_t)2 * kN * kD;  // elems per (2N,128)

  _Float16* fcA16   = (_Float16*)alloc((size_t)kD * kDin * 2);
  _Float16* fcP16   = (_Float16*)alloc((size_t)kD * kDin * 2);
  _Float16* inpaT   = (_Float16*)alloc((size_t)kD * kD * 2);
  _Float16* inapT   = (_Float16*)alloc((size_t)kD * kD * 2);
  _Float16* gcT     = (_Float16*)alloc((size_t)8 * kD * kD * 2);
  _Float16* afc16   = (_Float16*)alloc((size_t)2 * kD * kD * 2);
  _Float16* sain16  = (_Float16*)alloc((size_t)3 * kD * kD * 2);
  _Float16* saout16 = (_Float16*)alloc((size_t)kD * kD * 2);
  _Float16* cain16  = (_Float16*)alloc((size_t)3 * kD * kD * 2);
  _Float16* caout16 = (_Float16*)alloc((size_t)kD * kD * 2);
  _Float16* w116    = (_Float16*)alloc((size_t)kDff * kD * 2);
  _Float16* w216    = (_Float16*)alloc((size_t)kD * kDff * 2);
  _Float16* pred16  = (_Float16*)alloc((size_t)kOut * kDin * 2);
  _Float16* proj16  = (_Float16*)alloc(Nf * 2);
  float* curA  = (float*)alloc(Nf * 4);
  float* curP  = (float*)alloc(Nf * 4);
  float* pbuf0 = (float*)alloc(Nf * 4);
  float* pbuf1 = (float*)alloc(Nf * 4);
  float* hbuf[4];
  for (int l = 0; l < 4; l++) hbuf[l] = (float*)alloc(Nf * 4);
  float* msgA = (float*)alloc(Nf * 4);
  float* msgP = (float*)alloc(Nf * 4);
  float* degA = (float*)alloc((size_t)kN * 4);
  float* degP = (float*)alloc((size_t)kN * 4);
  float* colsumB = (float*)alloc(4 * kD * 4);
  float* betaB   = (float*)alloc(4 * 4);
  float* tgt32     = (float*)alloc(Nf2 * 4);
  _Float16* tgt16  = (_Float16*)alloc(Nf2 * 2);
  float* qkv       = (float*)alloc((size_t)2 * kN * 384 * 4);
  _Float16* ao16   = (_Float16*)alloc(Nf2 * 2);
  float* attnproj  = (float*)alloc(Nf2 * 4);   // also reused as FFN output
  float* x1        = (float*)alloc(Nf2 * 4);
  _Float16* x1_16  = (_Float16*)alloc(Nf2 * 2); // also reused as final x3_16
  float* x2        = (float*)alloc(Nf2 * 4);
  _Float16* x2_16  = (_Float16*)alloc(Nf2 * 2);

  // -------- weight conversion (f32 -> f16, transposed where used as x@W) ----
  cvt(stream, fcA_w, fcA16, (long long)kD * kDin);
  cvt(stream, fcP_w, fcP16, (long long)kD * kDin);
  cvtT(stream, in_w_pa, inpaT, kD, kD);
  cvtT(stream, in_w_ap, inapT, kD, kD);
  for (int i = 0; i < 8; i++)
    cvtT(stream, gc_w + (size_t)i * kD * kD, gcT + (size_t)i * kD * kD, kD, kD);
  cvt(stream, attn_fc_w, afc16, (long long)2 * kD * kD);
  cvt(stream, sa_in_w, sain16, (long long)3 * kD * kD);
  cvt(stream, sa_out_w, saout16, (long long)kD * kD);
  cvt(stream, ca_in_w, cain16, (long long)3 * kD * kD);
  cvt(stream, ca_out_w, caout16, (long long)kD * kD);
  cvt(stream, ff_w1, w116, (long long)kDff * kD);
  cvt(stream, ff_w2, w216, (long long)kD * kDff);
  cvt(stream, pred_w, pred16, (long long)kOut * kDin);

  // -------- degrees (1/(deg+1)), constant across layers --------
  hipMemsetAsync(degA, 0, (size_t)kN * 4, stream);
  hipMemsetAsync(degP, 0, (size_t)kN * 4, stream);
  k_count<<<(kE + 255) / 256, 256, 0, stream>>>(dst_pa, degA, kE);
  k_count<<<(kE + 255) / 256, 256, 0, stream>>>(dst_ap, degP, kE);
  k_invdeg<<<(kN + 255) / 256, 256, 0, stream>>>(degA, kN);
  k_invdeg<<<(kN + 255) / 256, 256, 0, stream>>>(degP, kN);

  // -------- input projections --------
  launch_gemm(stream, false, feat_A, kDin, fcA16, kDin, fcA_b, nullptr, nullptr,
              nullptr, 0, nullptr, 0, proj16, kD, kN, kD, kDin);
  launch_gemm(stream, true, proj16, kD, inpaT, kD, nullptr, nullptr, nullptr,
              nullptr, 0, curA, kD, nullptr, 0, kN, kD, kD);
  launch_gemm(stream, false, feat_P, kDin, fcP16, kDin, fcP_b, nullptr, nullptr,
              nullptr, 0, nullptr, 0, proj16, kD, kN, kD, kDin);
  launch_gemm(stream, true, proj16, kD, inapT, kD, nullptr, nullptr, nullptr,
              nullptr, 0, curP, kD, nullptr, 0, kN, kD, kD);

  // -------- graph conv layers --------
  const float* cA = curA;
  const float* cP = curP;
  const unsigned sgrid = (unsigned)(((long long)kE * 32 + 255) / 256);
  for (int l = 0; l < 4; l++) {
    hipMemsetAsync(msgA, 0, Nf * 4, stream);
    hipMemsetAsync(msgP, 0, Nf * 4, stream);
    k_scatter<<<sgrid, 256, 0, stream>>>(cP, src_pa, dst_pa, msgA, kE);
    k_scatter<<<sgrid, 256, 0, stream>>>(cA, src_ap, dst_ap, msgP, kE);
    float* newA = hbuf[l];
    float* newP = (l & 1) ? pbuf1 : pbuf0;
    launch_gemm(stream, false, msgA, kD, gcT + (size_t)(l * 2 + 0) * kD * kD, kD,
                gc_b + (l * 2 + 0) * kD, rand_f + (size_t)(2 * l + 0) * Nf, degA,
                prelu_a + (l * 2 + 0), 1, newA, kD, nullptr, 0, kN, kD, kD);
    launch_gemm(stream, false, msgP, kD, gcT + (size_t)(l * 2 + 1) * kD * kD, kD,
                gc_b + (l * 2 + 1) * kD, rand_f + (size_t)(2 * l + 1) * Nf, degP,
                prelu_a + (l * 2 + 1), 1, newP, kD, nullptr, 0, kN, kD, kD);
    cA = newA;
    cP = newP;
  }

  // -------- semantic attention: hA=[h0,h2], hP=[h1,h3] --------
  hipMemsetAsync(colsumB, 0, 4 * kD * 4, stream);
  for (int t = 0; t < 2; t++) {
    for (int m = 0; m < 2; m++) {
      const float* hsel = hbuf[(t == 0) ? (2 * m) : (2 * m + 1)];
      launch_gemm(stream, false, hsel, kD, afc16 + (size_t)t * kD * kD, kD,
                  attn_fc_b + t * kD, nullptr, nullptr, nullptr, 3, msgA, kD,
                  nullptr, 0, kN, kD, kD);  // tanh epilogue; msgA reused as tmp
      k_colsum<<<(kN + 255) / 256, 128, 0, stream>>>(
          msgA, colsumB + (t * 2 + m) * kD, kN);
    }
  }
  k_beta<<<1, 1, 0, stream>>>(colsumB, attn_vec, betaB);
  k_combine<<<(unsigned)((Nf + 255) / 256), 256, 0, stream>>>(
      hbuf[0], hbuf[2], betaB + 0, tgt32, tgt16, 0);
  k_combine<<<(unsigned)((Nf + 255) / 256), 256, 0, stream>>>(
      hbuf[1], hbuf[3], betaB + 2, tgt32, tgt16, 1);

  // -------- transformer decoder layer (post-LN) --------
  const int M2 = 2 * kN;
  // self-attention: qkv = tgt @ sa_in_w^T + b  -> (2N, 384)
  launch_gemm(stream, true, tgt16, kD, sain16, kD, sa_in_b, nullptr, nullptr,
              nullptr, 0, qkv, 384, nullptr, 0, M2, 384, kD);
  k_mha<<<(kN * 8 + 255) / 256, 256, 0, stream>>>(qkv, qkv + 128, qkv + 256, 384,
                                                  ao16, kN);
  launch_gemm(stream, true, ao16, kD, saout16, kD, sa_out_b, nullptr, nullptr,
              nullptr, 0, attnproj, kD, nullptr, 0, M2, kD, kD);
  k_ln<<<(M2 + 7) / 8, dim3(32, 8), 0, stream>>>(tgt32, attnproj, ln_g, ln_b, x1,
                                                 x1_16, M2);
  // cross-attention: q from x1, k/v from tgt
  launch_gemm(stream, true, x1_16, kD, cain16, kD, ca_in_b, nullptr, nullptr,
              nullptr, 0, qkv, 384, nullptr, 0, M2, kD, kD);
  launch_gemm(stream, true, tgt16, kD, cain16 + (size_t)128 * kD, kD,
              ca_in_b + 128, nullptr, nullptr, nullptr, 0, qkv + 128, 384,
              nullptr, 0, M2, kD, kD);
  launch_gemm(stream, true, tgt16, kD, cain16 + (size_t)256 * kD, kD,
              ca_in_b + 256, nullptr, nullptr, nullptr, 0, qkv + 256, 384,
              nullptr, 0, M2, kD, kD);
  k_mha<<<(kN * 8 + 255) / 256, 256, 0, stream>>>(qkv, qkv + 128, qkv + 256, 384,
                                                  ao16, kN);
  launch_gemm(stream, true, ao16, kD, caout16, kD, ca_out_b, nullptr, nullptr,
              nullptr, 0, attnproj, kD, nullptr, 0, M2, kD, kD);
  k_ln<<<(M2 + 7) / 8, dim3(32, 8), 0, stream>>>(x1, attnproj, ln_g + kD,
                                                 ln_b + kD, x2, x2_16, M2);
  // fused FFN (output into attnproj, which is free now)
  k_ffn<<<(M2 + 127) / 128, dim3(32, 8), 0, stream>>>(x2_16, w116, ff_b1, w216,
                                                      ff_b2, attnproj, M2);
  k_ln<<<(M2 + 7) / 8, dim3(32, 8), 0, stream>>>(x2, attnproj, ln_g + 2 * kD,
                                                 ln_b + 2 * kD, nullptr, x1_16,
                                                 M2);  // x1_16 reused as x3_16
  // prediction: view (N, 2, 128) as (N, 256)
  launch_gemm(stream, true, x1_16, kDin, pred16, kDin, pred_b, nullptr, nullptr,
              nullptr, 0, out, kOut, nullptr, 0, kN, kOut, kDin);
}
